// Repro_15075335209180
// MI455X (gfx1250) — compile-verified, matrix operand order checked by reading, not run on hardware
//
#include <hip/hip_runtime.h>
#include <hip/hip_bf16.h>
#include <math.h>

#define BN   4096
#define DIN  512
#define DD   128
#define NH   8
#define HD   16
#define MM   128
#define LL   129
#define DFF  512
#define SCALE_Q 0.25f

typedef __attribute__((ext_vector_type(16))) __bf16 v16bf;
typedef __attribute__((ext_vector_type(8)))  float  v8f;

union BFrag { v16bf v; unsigned int u[8]; unsigned short s[16]; };

__device__ __forceinline__ unsigned short f2bf(float f) {
  unsigned int u = __float_as_uint(f);
  u += 0x7FFFu + ((u >> 16) & 1u);           // round-to-nearest-even
  return (unsigned short)(u >> 16);
}
__device__ __forceinline__ float bf2f(unsigned short h) {
  return __uint_as_float(((unsigned int)h) << 16);
}

__device__ __forceinline__ float wave_sum(float v) {
  for (int o = 16; o > 0; o >>= 1) v += __shfl_xor(v, o, 32);
  return v;
}
__device__ __forceinline__ float wave_max(float v) {
  for (int o = 16; o > 0; o >>= 1) v = fmaxf(v, __shfl_xor(v, o, 32));
  return v;
}

// ---------------- fp32 -> bf16 convert ----------------
__global__ void cvt_bf16(const float* __restrict__ in, unsigned short* __restrict__ out, int n) {
  int i = blockIdx.x * 256 + threadIdx.x;
  if (i < n) out[i] = f2bf(in[i]);
}

// ---------------- bf16 WMMA GEMM: C = A(MxK) @ W(NxK)^T + bias [+extra | gelu] ---------
// EPI: 0 = bias, 1 = bias + extra[m,n], 2 = bias then GELU
template<int EPI>
__global__ void __launch_bounds__(256) gemm_bf16_wmma(
    const unsigned short* __restrict__ A,
    const unsigned short* __restrict__ W,
    const float* __restrict__ bias,
    const float* __restrict__ extra,
    float* __restrict__ C,
    int M, int N, int K)
{
  const int lane = threadIdx.x & 31;
  const int wave = threadIdx.x >> 5;
  const int m0   = blockIdx.y * 128 + wave * 16;
  const int n0   = blockIdx.x * 64;
  const int mrow = m0 + (lane & 15);
  const int kb0  = (lane >> 4) * 8;        // 16-bit A/B frag layout: lanes 0-15 K 0-7/16-23, lanes 16-31 K 8-15/24-31

  v8f z = {0.f,0.f,0.f,0.f,0.f,0.f,0.f,0.f};
  v8f acc[4] = {z, z, z, z};

  for (int k0 = 0; k0 < K; k0 += 32) {
    BFrag a;
    {
      const uint4* p0 = reinterpret_cast<const uint4*>(A + (size_t)mrow * K + k0 + kb0);
      const uint4* p1 = reinterpret_cast<const uint4*>(A + (size_t)mrow * K + k0 + kb0 + 16);
      uint4 t0 = *p0, t1 = *p1;
      a.u[0]=t0.x; a.u[1]=t0.y; a.u[2]=t0.z; a.u[3]=t0.w;
      a.u[4]=t1.x; a.u[5]=t1.y; a.u[6]=t1.z; a.u[7]=t1.w;
    }
#pragma unroll
    for (int j = 0; j < 4; ++j) {
      const int n = n0 + j * 16 + (lane & 15);
      BFrag b;
      const uint4* q0 = reinterpret_cast<const uint4*>(W + (size_t)n * K + k0 + kb0);
      const uint4* q1 = reinterpret_cast<const uint4*>(W + (size_t)n * K + k0 + kb0 + 16);
      uint4 t0 = *q0, t1 = *q1;
      b.u[0]=t0.x; b.u[1]=t0.y; b.u[2]=t0.z; b.u[3]=t0.w;
      b.u[4]=t1.x; b.u[5]=t1.y; b.u[6]=t1.z; b.u[7]=t1.w;
      acc[j] = __builtin_amdgcn_wmma_f32_16x16x32_bf16(
          false, a.v, false, b.v, (short)0, acc[j], false, false);
    }
  }

  const int mbase = m0 + (lane >> 4) * 8;   // C/D layout: lanes 0-15 M=v, lanes 16-31 M=v+8
  const int nn    = lane & 15;
#pragma unroll
  for (int j = 0; j < 4; ++j) {
    const int n = n0 + j * 16 + nn;
    const float bsv = bias[n];
#pragma unroll
    for (int v = 0; v < 8; ++v) {
      const int m = mbase + v;
      float val = acc[j][v] + bsv;
      if (EPI == 1) val += extra[(size_t)m * N + n];
      if (EPI == 2) val = 0.5f * val * (1.0f + erff(val * 0.70710678118654752f));
      C[(size_t)m * N + n] = val;
    }
  }
}

// ---------------- row LayerNorm -> bf16 ----------------
__global__ void __launch_bounds__(128) ln_rows(
    const float* __restrict__ in, const float* __restrict__ w, const float* __restrict__ b,
    unsigned short* __restrict__ out, int width)
{
  __shared__ float red[128];
  const int row = blockIdx.x;
  const int tid = threadIdx.x;
  const float* x = in + (size_t)row * width;
  float s = 0.f, s2 = 0.f;
  for (int c = tid; c < width; c += 128) { float v = x[c]; s += v; s2 += v * v; }
  red[tid] = s; __syncthreads();
  for (int st = 64; st > 0; st >>= 1) { if (tid < st) red[tid] += red[tid + st]; __syncthreads(); }
  float mean = red[0] / width; __syncthreads();
  red[tid] = s2; __syncthreads();
  for (int st = 64; st > 0; st >>= 1) { if (tid < st) red[tid] += red[tid + st]; __syncthreads(); }
  float inv = rsqrtf(red[0] / width - mean * mean + 1e-5f);
  for (int c = tid; c < width; c += 128)
    out[(size_t)row * width + c] = f2bf((x[c] - mean) * inv * w[c] + b[c]);
}

// ---------------- shared 128-wide LN stats ----------------
__device__ __forceinline__ void stats128(float v, float* red, int tid, float& mean, float& inv) {
  if (tid < 128) red[tid] = v;
  __syncthreads();
  for (int st = 64; st > 0; st >>= 1) { if (tid < st) red[tid] += red[tid + st]; __syncthreads(); }
  float s = red[0]; __syncthreads();
  if (tid < 128) red[tid] = v * v;
  __syncthreads();
  for (int st = 64; st > 0; st >>= 1) { if (tid < st) red[tid] += red[tid + st]; __syncthreads(); }
  float s2 = red[0]; __syncthreads();
  mean = s * (1.0f / 128.0f);
  inv  = rsqrtf(s2 * (1.0f / 128.0f) - mean * mean + 1e-5f);
}

// ---------------- fused attention (1 workgroup / batch element) ----------------
// Exploits single-query structure: scores = ln @ (SCALE*Wk^T qh), o = (p^T ln) @ Wv^T + bv.
__global__ void __launch_bounds__(256) attn_fused(
    const float* __restrict__ qin, const float* __restrict__ mem, const float* __restrict__ mask,
    const float* __restrict__ ln1w, const float* __restrict__ ln1b,
    const float* __restrict__ Wqkv, const float* __restrict__ bqkv,
    const float* __restrict__ Wout, const float* __restrict__ bout,
    const float* __restrict__ ln2w, const float* __restrict__ ln2b,
    const float* __restrict__ ln3w, const float* __restrict__ ln3b,
    float* __restrict__ h_out, unsigned short* __restrict__ h3_out)
{
  __shared__ unsigned short s_ln[LL * DD];   // ln1(kv_in), bf16, 33 KB
  __shared__ float s_qin[DD];
  __shared__ float s_q[DD];
  __shared__ float s_kproj[DD * NH];         // [c][h]
  __shared__ float s_hb[NH];
  __shared__ float s_p[NH * 132];
  __shared__ float s_u[NH * DD];             // [h][c]
  __shared__ float s_o[DD];
  __shared__ float s_t[DD];
  __shared__ float red[128];

  const int bid  = blockIdx.x;
  const int tid  = threadIdx.x;
  const int lane = tid & 31;
  const int wv   = tid >> 5;

  if (tid < DD) s_qin[tid] = qin[(size_t)bid * DD + tid];
  __syncthreads();

  // LN1 over the 129 kv rows (row 0 = q_in, rows 1..128 = mem), one wave per row
  for (int r = wv; r < LL; r += 8) {
    const float* src = (r == 0) ? s_qin : (mem + (size_t)bid * MM * DD + (size_t)(r - 1) * DD);
    if (r + 8 < LL)
      __builtin_prefetch(mem + (size_t)bid * MM * DD + (size_t)(r + 7) * DD + lane * 4, 0, 1);
    float vals[4]; float s = 0.f, s2 = 0.f;
#pragma unroll
    for (int i = 0; i < 4; ++i) { float v = src[lane + i * 32]; vals[i] = v; s += v; s2 += v * v; }
    s = wave_sum(s); s2 = wave_sum(s2);
    float mean = s * (1.0f / DD);
    float inv  = rsqrtf(s2 * (1.0f / DD) - mean * mean + 1e-5f);
#pragma unroll
    for (int i = 0; i < 4; ++i) {
      int c = lane + i * 32;
      s_ln[r * DD + c] = f2bf((vals[i] - mean) * inv * ln1w[c] + ln1b[c]);
    }
  }

  // q = q_in @ Wq^T + bq   (Wq = W_qkv rows 0..127)
  if (tid < DD) {
    float acc = bqkv[tid];
    const float* wr = Wqkv + (size_t)tid * DD;
    for (int c = 0; c < DD; ++c) acc += s_qin[c] * wr[c];
    s_q[tid] = acc;
  }
  __syncthreads();

  // Kproj[c][h] = SCALE * sum_d Wk[h*16+d][c] * q[h*16+d] ; head consts from bk
  for (int idx = tid; idx < DD * NH; idx += 256) {
    int c = idx >> 3, h = idx & 7;
    float acc = 0.f;
#pragma unroll
    for (int d = 0; d < HD; ++d)
      acc += Wqkv[(size_t)(DD + h * HD + d) * DD + c] * s_q[h * HD + d];
    s_kproj[idx] = SCALE_Q * acc;
  }
  if (tid < NH) {
    float acc = 0.f;
#pragma unroll
    for (int d = 0; d < HD; ++d) acc += s_q[tid * HD + d] * bqkv[DD + tid * HD + d];
    s_hb[tid] = SCALE_Q * acc;
  }
  __syncthreads();

  // scores + softmax: wave h handles head h over 129 positions
  {
    const int h = wv;
    float sc[5]; int cnt = 0; float mx = -3.0e38f;
    for (int l = lane; l < LL; l += 32) {
      float s = s_hb[h];
      for (int cc = 0; cc < DD; ++cc) {
        int c = (cc + lane * 4) & (DD - 1);
        s += bf2f(s_ln[l * DD + c]) * s_kproj[c * 8 + h];
      }
      if (l > 0) s += mask[(size_t)bid * MM + l - 1];
      sc[cnt++] = s; mx = fmaxf(mx, s);
    }
    mx = wave_max(mx);
    float sum = 0.f;
    for (int i = 0; i < cnt; ++i) { sc[i] = __expf(sc[i] - mx); sum += sc[i]; }
    sum = wave_sum(sum);
    float inv = 1.0f / sum;
    cnt = 0;
    for (int l = lane; l < LL; l += 32) s_p[h * 132 + l] = sc[cnt++] * inv;
  }
  __syncthreads();

  // u[h][c] = sum_l p[h][l] * ln[l][c]
  for (int idx = tid; idx < NH * DD; idx += 256) {
    int h = idx >> 7, c = idx & 127;
    float acc = 0.f;
    for (int l = 0; l < LL; ++l) acc += s_p[h * 132 + l] * bf2f(s_ln[l * DD + c]);
    s_u[idx] = acc;
  }
  __syncthreads();

  // o[n] = u[h] . Wv[n] + bv[n]  (Wv = W_qkv rows 256..383)
  if (tid < DD) {
    int h = tid >> 4;
    float acc = bqkv[2 * DD + tid];
    const float* wr = Wqkv + (size_t)(2 * DD + tid) * DD;
    for (int c = 0; c < DD; ++c) acc += s_u[h * DD + c] * wr[c];
    s_o[tid] = acc;
  }
  __syncthreads();

  // out projection
  if (tid < DD) {
    float acc = bout[tid];
    const float* wr = Wout + (size_t)tid * DD;
    for (int c = 0; c < DD; ++c) acc += s_o[c] * wr[c];
    s_t[tid] = acc;
  }
  __syncthreads();

  // ln2 -> residual with q_in -> h ; fused ln3 -> bf16
  float myv = (tid < DD) ? s_t[tid] : 0.f;
  float mean, inv;
  stats128(myv, red, tid, mean, inv);
  float hval = 0.f;
  if (tid < DD) {
    hval = (myv - mean) * inv * ln2w[tid] + ln2b[tid] + s_qin[tid];
    h_out[(size_t)bid * DD + tid] = hval;
  }
  stats128(hval, red, tid, mean, inv);
  if (tid < DD)
    h3_out[(size_t)bid * DD + tid] = f2bf((hval - mean) * inv * ln3w[tid] + ln3b[tid]);
}

// ---------------- launch ----------------
extern "C" void kernel_launch(void* const* d_in, const int* in_sizes, int n_in,
                              void* d_out, int out_size, void* d_ws, size_t ws_size,
                              hipStream_t stream) {
  (void)in_sizes; (void)n_in; (void)out_size; (void)ws_size;
  const float* x    = (const float*)d_in[0];
  const float* W_in = (const float*)d_in[1];
  const float* b_in = (const float*)d_in[2];
  const float* pos  = (const float*)d_in[3];
  const float* mask = (const float*)d_in[4];
  const float* mem  = (const float*)d_in[5];
  const float* ln1w = (const float*)d_in[6];
  const float* ln1b = (const float*)d_in[7];
  const float* Wqkv = (const float*)d_in[8];
  const float* bqkv = (const float*)d_in[9];
  const float* Wout = (const float*)d_in[10];
  const float* bout = (const float*)d_in[11];
  const float* ln2w = (const float*)d_in[12];
  const float* ln2b = (const float*)d_in[13];
  const float* ln3w = (const float*)d_in[14];
  const float* ln3b = (const float*)d_in[15];
  const float* Wff1 = (const float*)d_in[16];
  const float* bff1 = (const float*)d_in[17];
  const float* ln4w = (const float*)d_in[18];
  const float* ln4b = (const float*)d_in[19];
  const float* Wff2 = (const float*)d_in[20];
  const float* bff2 = (const float*)d_in[21];
  const float* ln5w = (const float*)d_in[22];
  const float* ln5b = (const float*)d_in[23];
  const float* Whead= (const float*)d_in[24];
  const float* bhead= (const float*)d_in[25];
  float* out = (float*)d_out;

  char* ws = (char*)d_ws;
  size_t off = 0;
  auto alloc = [&](size_t bytes) -> void* {
    void* p = ws + off; off += (bytes + 255) & ~(size_t)255; return p;
  };
  unsigned short* x_bf    = (unsigned short*)alloc((size_t)BN * DIN * 2);
  unsigned short* Win_bf  = (unsigned short*)alloc((size_t)DD * DIN * 2);
  unsigned short* Wff1_bf = (unsigned short*)alloc((size_t)DFF * DD * 2);
  unsigned short* Wff2_bf = (unsigned short*)alloc((size_t)DD * DFF * 2);
  unsigned short* Whead_bf= (unsigned short*)alloc((size_t)DIN * DD * 2);
  float*          qin     = (float*)alloc((size_t)BN * DD * 4);
  float*          h       = (float*)alloc((size_t)BN * DD * 4);
  unsigned short* h3      = (unsigned short*)alloc((size_t)BN * DD * 2);
  float*          f       = (float*)alloc((size_t)BN * DFF * 4);
  unsigned short* g       = (unsigned short*)alloc((size_t)BN * DFF * 2);
  float*          t       = (float*)alloc((size_t)BN * DD * 4);
  unsigned short* t5      = (unsigned short*)alloc((size_t)BN * DD * 2);

  auto cvt = [&](const float* src, unsigned short* dst, int n) {
    cvt_bf16<<<(n + 255) / 256, 256, 0, stream>>>(src, dst, n);
  };
  cvt(x,     x_bf,     BN * DIN);
  cvt(W_in,  Win_bf,   DD * DIN);
  cvt(Wff1,  Wff1_bf,  DFF * DD);
  cvt(Wff2,  Wff2_bf,  DD * DFF);
  cvt(Whead, Whead_bf, DIN * DD);

  // q_in = x @ W_in^T + b_in + pos
  gemm_bf16_wmma<1><<<dim3(DD / 64, BN / 128), 256, 0, stream>>>(
      x_bf, Win_bf, b_in, pos, qin, BN, DD, DIN);
  // fused ln1 + attention + out-proj + ln2 + residual + ln3
  attn_fused<<<BN, 256, 0, stream>>>(qin, mem, mask, ln1w, ln1b, Wqkv, bqkv,
                                     Wout, bout, ln2w, ln2b, ln3w, ln3b, h, h3);
  // f = gelu(ln3(h) @ W_ff1^T + b_ff1)
  gemm_bf16_wmma<2><<<dim3(DFF / 64, BN / 128), 256, 0, stream>>>(
      h3, Wff1_bf, bff1, nullptr, f, BN, DFF, DD);
  ln_rows<<<BN, 128, 0, stream>>>(f, ln4w, ln4b, g, DFF);
  // t = ln4(f) @ W_ff2^T + b_ff2 + h
  gemm_bf16_wmma<1><<<dim3(DD / 64, BN / 128), 256, 0, stream>>>(
      g, Wff2_bf, bff2, h, t, BN, DD, DFF);
  ln_rows<<<BN, 128, 0, stream>>>(t, ln5w, ln5b, t5, DD);
  // out = ln5(t) @ W_head^T + b_head
  gemm_bf16_wmma<0><<<dim3(DIN / 64, BN / 128), 256, 0, stream>>>(
      t5, Whead_bf, bhead, nullptr, out, BN, DIN, DD);
}
